// EncoderDecoderRetrievalModel_40750649704579
// MI455X (gfx1250) — matrix operand order repaired and checked by reading, not compiled, not run on hardware
//
#include <hip/hip_runtime.h>

typedef __bf16 bf16;
typedef __attribute__((ext_vector_type(16))) __bf16 v16bf;
typedef __attribute__((ext_vector_type(8)))  __bf16 v8bf;
typedef __attribute__((ext_vector_type(8)))  float  v8f;

// ---- model dims (fixed by reference) ----
#define BB 32
#define SID_C 4
#define DD 512
#define AA 512
#define NHC 8
#define HD 64
#define LLAYERS 4
#define VV 2048
#define DFF_C 1024
#define N_TOK_C 512
#define S_ENC_C 513
#define U_HASH_C 2000
#define NEGM (-1.0e9f)

__device__ __forceinline__ v16bf cat8(v8bf lo, v8bf hi) {
    return __builtin_shufflevector(lo, hi, 0,1,2,3,4,5,6,7,8,9,10,11,12,13,14,15);
}

// Async global->LDS 16B copy (CDNA5 TDM-lite path, ASYNCcnt-tracked).
// LDS operand takes the wave-relative byte offset = low 32 bits of the
// generic pointer (LDS aperture keeps the offset in addr[31:0]).
__device__ __forceinline__ void async_ld_b128(const void* g, void* l) {
    unsigned lds = (unsigned)(size_t)l;
    unsigned long long ga = (unsigned long long)(size_t)g;
    asm volatile("global_load_async_to_lds_b128 %0, %1, off"
                 :: "v"(lds), "v"(ga) : "memory");
}
__device__ __forceinline__ void wait_async0() {
    asm volatile("s_wait_asynccnt 0" ::: "memory");
}

// =====================================================================
// Weight pre-pass: out[l][n][k] = (bf16) in[l][k][n]   (LDS 32x33 tile)
// =====================================================================
__global__ __launch_bounds__(256)
void transpose_cvt_kernel(const float* __restrict__ in, bf16* __restrict__ out,
                          int K, int N)
{
    __shared__ float t[32][33];
    const float* inl = in + (size_t)blockIdx.z * K * N;
    bf16* outl = out + (size_t)blockIdx.z * K * N;
    int k0 = blockIdx.x * 32;
    int n0 = blockIdx.y * 32;
    int tx = threadIdx.x & 31;
    int ty = threadIdx.x >> 5;
    #pragma unroll
    for (int i = 0; i < 4; ++i) {
        int k = k0 + ty + 8 * i;
        int n = n0 + tx;
        t[ty + 8 * i][tx] = (k < K && n < N) ? inl[(size_t)k * N + n] : 0.0f;
    }
    __syncthreads();
    #pragma unroll
    for (int i = 0; i < 4; ++i) {
        int n = n0 + ty + 8 * i;
        int k = k0 + tx;
        if (n < N && k < K) outl[(size_t)n * K + k] = (bf16)t[tx][ty + 8 * i];
    }
}

// =====================================================================
// GEMM: C[M,N] = act( A(bf16)[M,K] @ Bt(bf16)[N,K]^T + res )
// 128x64 C tile / 256 threads (8 waves); wave owns a 16x64 strip
// (4 WMMA accumulators). Double-buffered LDS fed by async b128 copies:
// prefetch of k-step i+1 overlaps the 4 WMMAs of k-step i.
// OOB rows handled by clamping (clamped rows feed never-stored C rows).
// =====================================================================
__global__ __launch_bounds__(256)
void gemm_kernel(const bf16* __restrict__ A, int lda,
                 const bf16* __restrict__ Bt,          // [N][K]
                 const float* __restrict__ res,
                 float* __restrict__ Cf, bf16* __restrict__ Cb,
                 int M, int N, int K, int act)
{
    __shared__ __align__(16) bf16 Ash[2][128][32];
    __shared__ __align__(32) bf16 Bsh[2][64][32];
    const int tid  = threadIdx.x;
    const int lane = tid & 31;
    const int wv   = tid >> 5;
    const int hl   = lane >> 4;
    const int l16  = lane & 15;
    const int bm   = blockIdx.y * 128;
    const int bn   = blockIdx.x * 64;

    // staging map: thread -> (row, 8-elem k group); rows clamped
    const int r0   = tid >> 2;        // 0..63
    const int r1   = r0 + 64;         // 64..127
    const int kg8  = (tid & 3) * 8;
    const int gm0  = min(bm + r0, M - 1);
    const int gm1  = min(bm + r1, M - 1);
    const int gn   = min(bn + r0, N - 1);
    const bf16* pa0 = A  + (size_t)gm0 * lda + kg8;
    const bf16* pa1 = A  + (size_t)gm1 * lda + kg8;
    const bf16* pb  = Bt + (size_t)gn  * K   + kg8;

    v8f acc[4] = {};
    const int nk = K >> 5;            // K is a multiple of 32

    // prologue: prefetch k-step 0 into buffer 0
    async_ld_b128(pa0, &Ash[0][r0][kg8]);
    async_ld_b128(pa1, &Ash[0][r1][kg8]);
    async_ld_b128(pb,  &Bsh[0][r0][kg8]);

    for (int it = 0; it < nk; ++it) {
        const int buf = it & 1;
        wait_async0();                // own batch `it` has landed in LDS
        __syncthreads();              // everyone's batch visible; prev buf free
        if (it + 1 < nk) {            // prefetch batch it+1 into other buffer
            const int ko = (it + 1) * 32;
            async_ld_b128(pa0 + ko, &Ash[buf ^ 1][r0][kg8]);
            async_ld_b128(pa1 + ko, &Ash[buf ^ 1][r1][kg8]);
            async_ld_b128(pb  + ko, &Bsh[buf ^ 1][r0][kg8]);
        }
        // compute on current buffer
        const int am = wv * 16 + l16;
        v8bf alo = *(const v8bf*)&Ash[buf][am][16 * hl];
        v8bf ahi = *(const v8bf*)&Ash[buf][am][16 * hl + 8];
        v16bf af = cat8(alo, ahi);
        #pragma unroll
        for (int j = 0; j < 4; ++j) {
            v16bf bf = *(const v16bf*)&Bsh[buf][j * 16 + l16][16 * hl];
            acc[j] = __builtin_amdgcn_wmma_f32_16x16x32_bf16(false, af, false, bf, (short)0, acc[j], false, false);
        }
    }

    #pragma unroll
    for (int j = 0; j < 4; ++j) {
        #pragma unroll
        for (int r = 0; r < 8; ++r) {
            int gm = bm + wv * 16 + r + 8 * hl;
            int gnn = bn + j * 16 + l16;
            if (gm >= M || gnn >= N) continue;
            float v = acc[j][r];
            if (res) v += res[(size_t)gm * N + gnn];
            if (act == 1) {
                float x = v;
                v = 0.5f * x * (1.0f + tanhf(0.7978845608028654f * (x + 0.044715f * x * x * x)));
            }
            if (Cf) Cf[(size_t)gm * N + gnn] = v;
            if (Cb) Cb[(size_t)gm * N + gnn] = (bf16)v;
        }
    }
}

// =====================================================================
// Flash attention with WMMA for QK^T and PV. 4 waves/block, 16 q/wave.
// Q/K/V/O: [b, s, h*64+d] bf16.  kmask: [b, Sk] int (or null).
// OOB keys are masked by INDEX to -3e38 before softmax, so clamped
// (finite) K/V loads contribute exactly zero.
// =====================================================================
__global__ __launch_bounds__(128)
void attn_kernel(const bf16* __restrict__ Q, const bf16* __restrict__ Kt,
                 const bf16* __restrict__ Vt, bf16* __restrict__ O,
                 const int* __restrict__ kmask, int Sq, int Sk,
                 int causal, float scale)
{
    __shared__ __align__(32) bf16 Ksh[32][64];      // [key][d]
    __shared__ __align__(32) bf16 Vsh[64][32];      // transposed: [d][key]
    __shared__ __align__(16) bf16 Psh[4][16][32];   // per-wave [q][key]

    const int tid  = threadIdx.x;
    const int lane = tid & 31;
    const int wv   = tid >> 5;
    const int hl   = lane >> 4;
    const int l16  = lane & 15;
    const int b    = blockIdx.x / NHC;
    const int h    = blockIdx.x % NHC;
    const int q0r  = (blockIdx.y * 4 + wv) * 16;

    v16bf qf0, qf1;
    {
        int qrow = min(q0r + l16, Sq - 1);
        const bf16* qp = Q + ((size_t)(b * Sq + qrow) * (NHC * HD)) + h * HD;
        v8bf a0 = *(const v8bf*)(qp + 16 * hl);
        v8bf a1 = *(const v8bf*)(qp + 8 + 16 * hl);
        v8bf b0 = *(const v8bf*)(qp + 32 + 16 * hl);
        v8bf b1 = *(const v8bf*)(qp + 40 + 16 * hl);
        qf0 = cat8(a0, a1);
        qf1 = cat8(b0, b1);
    }

    v8f accO[4] = {};
    float mrun[8], lrun[8];
    #pragma unroll
    for (int r = 0; r < 8; ++r) { mrun[r] = -3.0e38f; lrun[r] = 0.0f; }

    for (int c0 = 0; c0 < Sk; c0 += 32) {
        #pragma unroll
        for (int c = 0; c < 2; ++c) {
            int ch  = tid + c * 128;
            int key = ch >> 3;
            int dg  = ch & 7;
            int gk  = min(c0 + key, Sk - 1);
            size_t base = ((size_t)(b * Sk + gk) * (NHC * HD)) + h * HD + dg * 8;
            v8bf vk = *(const v8bf*)(Kt + base);
            v8bf vv = *(const v8bf*)(Vt + base);
            *(v8bf*)&Ksh[key][dg * 8] = vk;
            #pragma unroll
            for (int e = 0; e < 8; ++e) Vsh[dg * 8 + e][key] = vv[e];
        }
        __syncthreads();

        v8f sc[2];
        #pragma unroll
        for (int kt = 0; kt < 2; ++kt) {
            v16bf bf0 = *(const v16bf*)&Ksh[kt * 16 + l16][16 * hl];
            v16bf bf1 = *(const v16bf*)&Ksh[kt * 16 + l16][32 + 16 * hl];
            v8f s = {};
            s = __builtin_amdgcn_wmma_f32_16x16x32_bf16(false, qf0, false, bf0, (short)0, s, false, false);
            s = __builtin_amdgcn_wmma_f32_16x16x32_bf16(false, qf1, false, bf1, (short)0, s, false, false);
            sc[kt] = s;
        }

        const int k0i = c0 + l16, k1i = c0 + 16 + l16;
        const bool pad0 = (k0i >= Sk), pad1 = (k1i >= Sk);
        const bool msk0 = !pad0 && kmask && !kmask[b * Sk + k0i];
        const bool msk1 = !pad1 && kmask && !kmask[b * Sk + k1i];
        #pragma unroll
        for (int r = 0; r < 8; ++r) {
            int qrow = q0r + r + 8 * hl;
            float s0 = sc[0][r] * scale;
            float s1 = sc[1][r] * scale;
            if (pad0)                                s0 = -3.0e38f;
            else if ((causal && k0i > qrow) || msk0) s0 = NEGM;
            if (pad1)                                s1 = -3.0e38f;
            else if ((causal && k1i > qrow) || msk1) s1 = NEGM;

            float mx = fmaxf(s0, s1);
            #pragma unroll
            for (int off = 8; off; off >>= 1) mx = fmaxf(mx, __shfl_xor(mx, off, 32));
            float mnew = fmaxf(mrun[r], mx);
            float corr = __expf(mrun[r] - mnew);
            float p0 = __expf(s0 - mnew);
            float p1 = __expf(s1 - mnew);
            float rs = p0 + p1;
            #pragma unroll
            for (int off = 8; off; off >>= 1) rs += __shfl_xor(rs, off, 32);
            lrun[r] = lrun[r] * corr + rs;
            mrun[r] = mnew;
            #pragma unroll
            for (int ng = 0; ng < 4; ++ng) accO[ng][r] *= corr;
            Psh[wv][r + 8 * hl][l16]      = (bf16)p0;
            Psh[wv][r + 8 * hl][16 + l16] = (bf16)p1;
        }
        __syncthreads();

        v8bf plo = *(const v8bf*)&Psh[wv][l16][16 * hl];
        v8bf phi = *(const v8bf*)&Psh[wv][l16][16 * hl + 8];
        v16bf pf = cat8(plo, phi);
        #pragma unroll
        for (int ng = 0; ng < 4; ++ng) {
            v16bf vf = *(const v16bf*)&Vsh[ng * 16 + l16][16 * hl];
            accO[ng] = __builtin_amdgcn_wmma_f32_16x16x32_bf16(false, pf, false, vf, (short)0, accO[ng], false, false);
        }
        __syncthreads();
    }

    #pragma unroll
    for (int r = 0; r < 8; ++r) {
        int qrow = q0r + r + 8 * hl;
        if (qrow >= Sq) continue;
        float inv = 1.0f / lrun[r];
        #pragma unroll
        for (int ng = 0; ng < 4; ++ng) {
            int d = ng * 16 + l16;
            O[((size_t)(b * Sq + qrow) * (NHC * HD)) + h * HD + d] = (bf16)(accO[ng][r] * inv);
        }
    }
}

// =====================================================================
// RMS norm (row-wise) fp32 -> bf16
// =====================================================================
__global__ __launch_bounds__(256)
void rmsnorm_kernel(const float* __restrict__ x, const float* __restrict__ w,
                    bf16* __restrict__ out, int dim)
{
    __shared__ float red[256];
    int row = blockIdx.x;
    const float* xr = x + (size_t)row * dim;
    float acc = 0.f;
    for (int i = threadIdx.x; i < dim; i += 256) { float v = xr[i]; acc += v * v; }
    red[threadIdx.x] = acc;
    __syncthreads();
    for (int s = 128; s > 0; s >>= 1) {
        if (threadIdx.x < (unsigned)s) red[threadIdx.x] += red[threadIdx.x + s];
        __syncthreads();
    }
    float scale = rsqrtf(red[0] / (float)dim + 1e-6f);
    for (int i = threadIdx.x; i < dim; i += 256)
        out[(size_t)row * dim + i] = (bf16)(xr[i] * w[i] * scale);
}

// =====================================================================
// Embedding / mask / convert kernels
// =====================================================================
__global__ void embed_enc_kernel(const int* __restrict__ user_ids,
                                 const int* __restrict__ sem_ids,
                                 const float* __restrict__ table,
                                 const float* __restrict__ utab,
                                 const float* __restrict__ wpe,
                                 float* __restrict__ x)
{
    size_t i = (size_t)blockIdx.x * blockDim.x + threadIdx.x;
    size_t total = (size_t)BB * S_ENC_C * DD;
    if (i >= total) return;
    int d = (int)(i % DD);
    int p = (int)((i / DD) % S_ENC_C);
    int b = (int)(i / ((size_t)DD * S_ENC_C));
    float v;
    if (p == 0) {
        int u = user_ids[b] % U_HASH_C;
        v = utab[(size_t)u * DD + d];
    } else {
        int j = p - 1;
        int sid = sem_ids[b * N_TOK_C + j] + (j % SID_C) * VV;
        v = table[(size_t)sid * DD + d] + wpe[(size_t)j * DD + d];
    }
    x[i] = v;
}

__global__ void embed_dec_kernel(const int* __restrict__ sem_ids_fut,
                                 const float* __restrict__ table,
                                 const float* __restrict__ tte,
                                 const float* __restrict__ bos,
                                 float* __restrict__ y)
{
    int i = blockIdx.x * blockDim.x + threadIdx.x;
    if (i >= BB * SID_C * DD) return;
    int d = i % DD;
    int t = (i / DD) % SID_C;
    int b = i / (DD * SID_C);
    float v;
    if (t == 0) v = bos[d];
    else {
        int sid = sem_ids_fut[b * SID_C + (t - 1)] + (t - 1) * VV;
        v = table[(size_t)sid * DD + d];
    }
    y[i] = v + tte[t * DD + d];
}

__global__ void build_mask_kernel(const int* __restrict__ item_lengths, int* __restrict__ mask)
{
    int i = blockIdx.x * blockDim.x + threadIdx.x;
    if (i >= BB * S_ENC_C) return;
    int p = i % S_ENC_C;
    int b = i / S_ENC_C;
    mask[i] = (p == 0) ? 1 : ((((p - 1) / SID_C) < item_lengths[b]) ? 1 : 0);
}

__global__ void cvt_bf16_kernel(const float* __restrict__ in, bf16* __restrict__ out, size_t n)
{
    size_t i = (size_t)blockIdx.x * blockDim.x + threadIdx.x;
    if (i < n) out[i] = (bf16)in[i];
}

// =====================================================================
// Host orchestration
// Input order assumed = setup_inputs() insertion order, nested dicts
// flattened in written key order.
// =====================================================================
extern "C" void kernel_launch(void* const* d_in, const int* in_sizes, int n_in,
                              void* d_out, int out_size, void* d_ws, size_t ws_size,
                              hipStream_t stream)
{
    (void)in_sizes; (void)n_in; (void)out_size; (void)ws_size;

    const int* user_ids    = (const int*)d_in[0];
    const int* sem_ids     = (const int*)d_in[1];
    const int* sem_ids_fut = (const int*)d_in[2];
    const int* item_len    = (const int*)d_in[3];
    const float* table     = (const float*)d_in[4];
    const float* utab      = (const float*)d_in[5];
    const float* wpe       = (const float*)d_in[6];
    const float* tte       = (const float*)d_in[7];
    const float* bos       = (const float*)d_in[8];
    const float* norm_w    = (const float*)d_in[9];
    const float* norm_cw   = (const float*)d_in[10];
    const float* in_proj   = (const float*)d_in[11];
    const float* in_projc  = (const float*)d_in[12];
    const float* out_proj  = (const float*)d_in[13];
    const float* e_ln1 = (const float*)d_in[14];
    const float* e_ln2 = (const float*)d_in[15];
    const float* e_Wq  = (const float*)d_in[16];
    const float* e_Wk  = (const float*)d_in[17];
    const float* e_Wv  = (const float*)d_in[18];
    const float* e_Wo  = (const float*)d_in[19];
    const float* e_W1  = (const float*)d_in[20];
    const float* e_W2  = (const float*)d_in[21];
    const float* g_ln1 = (const float*)d_in[22];
    const float* g_lnc = (const float*)d_in[23];
    const float* g_ln2 = (const float*)d_in[24];
    const float* g_Wq  = (const float*)d_in[25];
    const float* g_Wk  = (const float*)d_in[26];
    const float* g_Wv  = (const float*)d_in[27];
    const float* g_Wo  = (const float*)d_in[28];
    const float* g_Cq  = (const float*)d_in[29];
    const float* g_Ck  = (const float*)d_in[30];
    const float* g_Cv  = (const float*)d_in[31];
    const float* g_Co  = (const float*)d_in[32];
    const float* g_W1  = (const float*)d_in[33];
    const float* g_W2  = (const float*)d_in[34];

    const int Menc = BB * S_ENC_C;   // 16416
    const int Mdec = BB * SID_C;     // 128

    char* wp = (char*)d_ws;
    size_t off = 0;
    auto carve = [&](size_t bytes) -> void* {
        void* p = wp + off;
        off += (bytes + 255) & ~(size_t)255;
        return p;
    };
    void* big   = carve((size_t)Menc * DFF_C * sizeof(bf16)); // x_enc f32 / tffn bf16
    float* x_enc = (float*)big;
    bf16*  tffn  = (bf16*)big;
    float* hbuf = (float*)carve((size_t)Menc * AA * sizeof(float));
    bf16*  zn   = (bf16*)carve((size_t)Menc * AA * sizeof(bf16));
    bf16*  qb   = (bf16*)carve((size_t)Menc * AA * sizeof(bf16));
    bf16*  kb   = (bf16*)carve((size_t)Menc * AA * sizeof(bf16));
    bf16*  vb   = (bf16*)carve((size_t)Menc * AA * sizeof(bf16));
    bf16*  ob   = (bf16*)carve((size_t)Menc * AA * sizeof(bf16));
    bf16*  ctxb = (bf16*)carve((size_t)Menc * AA * sizeof(bf16));
    int*   maskw= (int*)carve((size_t)Menc * sizeof(int));
    float* y0   = (float*)carve((size_t)Mdec * DD * sizeof(float));
    float* ybuf = (float*)carve((size_t)Mdec * AA * sizeof(float));
    bf16*  dzn  = (bf16*)carve((size_t)Mdec * AA * sizeof(bf16));
    bf16*  dq   = (bf16*)carve((size_t)Mdec * AA * sizeof(bf16));
    bf16*  dk   = (bf16*)carve((size_t)Mdec * AA * sizeof(bf16));
    bf16*  dv   = (bf16*)carve((size_t)Mdec * AA * sizeof(bf16));
    bf16*  dob  = (bf16*)carve((size_t)Mdec * AA * sizeof(bf16));
    bf16*  dff  = (bf16*)carve((size_t)Mdec * DFF_C * sizeof(bf16));
    bf16*  ybf  = (bf16*)carve((size_t)Mdec * AA * sizeof(bf16));
    const size_t SZ_AA = (size_t)AA * AA;
    bf16* t_in  = (bf16*)carve(SZ_AA * sizeof(bf16));
    bf16* t_inc = (bf16*)carve(SZ_AA * sizeof(bf16));
    bf16* t_out = (bf16*)carve((size_t)AA * VV * sizeof(bf16));
    bf16* te_Wq = (bf16*)carve(LLAYERS * SZ_AA * sizeof(bf16));
    bf16* te_Wk = (bf16*)carve(LLAYERS * SZ_AA * sizeof(bf16));
    bf16* te_Wv = (bf16*)carve(LLAYERS * SZ_AA * sizeof(bf16));
    bf16* te_Wo = (bf16*)carve(LLAYERS * SZ_AA * sizeof(bf16));
    bf16* te_W1 = (bf16*)carve((size_t)LLAYERS * AA * DFF_C * sizeof(bf16));
    bf16* te_W2 = (bf16*)carve((size_t)LLAYERS * DFF_C * AA * sizeof(bf16));
    bf16* tg_Wq = (bf16*)carve(LLAYERS * SZ_AA * sizeof(bf16));
    bf16* tg_Wk = (bf16*)carve(LLAYERS * SZ_AA * sizeof(bf16));
    bf16* tg_Wv = (bf16*)carve(LLAYERS * SZ_AA * sizeof(bf16));
    bf16* tg_Wo = (bf16*)carve(LLAYERS * SZ_AA * sizeof(bf16));
    bf16* tg_Cq = (bf16*)carve(LLAYERS * SZ_AA * sizeof(bf16));
    bf16* tg_Ck = (bf16*)carve(LLAYERS * SZ_AA * sizeof(bf16));
    bf16* tg_Cv = (bf16*)carve(LLAYERS * SZ_AA * sizeof(bf16));
    bf16* tg_Co = (bf16*)carve(LLAYERS * SZ_AA * sizeof(bf16));
    bf16* tg_W1 = (bf16*)carve((size_t)LLAYERS * AA * DFF_C * sizeof(bf16));
    bf16* tg_W2 = (bf16*)carve((size_t)LLAYERS * DFF_C * AA * sizeof(bf16));

    auto tcvt = [&](const float* w, bf16* o, int L, int Kd, int Nd) {
        dim3 g(Kd / 32, Nd / 32, L);
        transpose_cvt_kernel<<<g, 256, 0, stream>>>(w, o, Kd, Nd);
    };
    auto gemm = [&](const bf16* A, int lda, const bf16* Bt,
                    const float* res, float* Cf, bf16* Cb,
                    int M, int N, int K, int act) {
        dim3 g((N + 63) / 64, (M + 127) / 128);
        gemm_kernel<<<g, 256, 0, stream>>>(A, lda, Bt, res, Cf, Cb, M, N, K, act);
    };
    auto rms = [&](const float* x, const float* w, bf16* o, int rows, int dim) {
        rmsnorm_kernel<<<rows, 256, 0, stream>>>(x, w, o, dim);
    };
    auto attn = [&](const bf16* Q, const bf16* K, const bf16* V, bf16* O,
                    const int* km, int Sq, int Sk, int causal) {
        dim3 g(BB * NHC, (Sq + 63) / 64);
        attn_kernel<<<g, 128, 0, stream>>>(Q, K, V, O, km, Sq, Sk, causal, 0.125f);
    };

    // ---- weight pre-pass (transpose + bf16) ----
    tcvt(in_proj,  t_in,  1, DD, AA);
    tcvt(in_projc, t_inc, 1, DD, AA);
    tcvt(out_proj, t_out, 1, AA, VV);
    tcvt(e_Wq, te_Wq, LLAYERS, AA, AA);
    tcvt(e_Wk, te_Wk, LLAYERS, AA, AA);
    tcvt(e_Wv, te_Wv, LLAYERS, AA, AA);
    tcvt(e_Wo, te_Wo, LLAYERS, AA, AA);
    tcvt(e_W1, te_W1, LLAYERS, AA, DFF_C);
    tcvt(e_W2, te_W2, LLAYERS, DFF_C, AA);
    tcvt(g_Wq, tg_Wq, LLAYERS, AA, AA);
    tcvt(g_Wk, tg_Wk, LLAYERS, AA, AA);
    tcvt(g_Wv, tg_Wv, LLAYERS, AA, AA);
    tcvt(g_Wo, tg_Wo, LLAYERS, AA, AA);
    tcvt(g_Cq, tg_Cq, LLAYERS, AA, AA);
    tcvt(g_Ck, tg_Ck, LLAYERS, AA, AA);
    tcvt(g_Cv, tg_Cv, LLAYERS, AA, AA);
    tcvt(g_Co, tg_Co, LLAYERS, AA, AA);
    tcvt(g_W1, tg_W1, LLAYERS, AA, DFF_C);
    tcvt(g_W2, tg_W2, LLAYERS, DFF_C, AA);

    // ---- encoder front ----
    build_mask_kernel<<<(Menc + 255) / 256, 256, 0, stream>>>(item_len, maskw);
    {
        size_t tot = (size_t)Menc * DD;
        embed_enc_kernel<<<(unsigned)((tot + 255) / 256), 256, 0, stream>>>(
            user_ids, sem_ids, table, utab, wpe, x_enc);
    }
    rms(x_enc, norm_w, zn, Menc, DD);
    gemm(zn, DD, t_in, nullptr, hbuf, nullptr, Menc, AA, DD, 0);

    // ---- encoder layers ----
    for (int l = 0; l < LLAYERS; ++l) {
        const size_t wo = (size_t)l * SZ_AA;
        const size_t wf = (size_t)l * AA * DFF_C;
        rms(hbuf, e_ln1 + (size_t)l * AA, zn, Menc, AA);
        gemm(zn, AA, te_Wq + wo, nullptr, nullptr, qb, Menc, AA, AA, 0);
        gemm(zn, AA, te_Wk + wo, nullptr, nullptr, kb, Menc, AA, AA, 0);
        gemm(zn, AA, te_Wv + wo, nullptr, nullptr, vb, Menc, AA, AA, 0);
        attn(qb, kb, vb, ob, maskw, S_ENC_C, S_ENC_C, 0);
        gemm(ob, AA, te_Wo + wo, hbuf, hbuf, nullptr, Menc, AA, AA, 0);
        rms(hbuf, e_ln2 + (size_t)l * AA, zn, Menc, AA);
        gemm(zn, AA, te_W1 + wf, nullptr, nullptr, tffn, Menc, DFF_C, AA, 1);
        gemm(tffn, DFF_C, te_W2 + wf, hbuf, hbuf, nullptr, Menc, AA, DFF_C, 0);
    }

    // ---- context to bf16 for cross-attn projections ----
    {
        size_t n = (size_t)Menc * AA;
        cvt_bf16_kernel<<<(unsigned)((n + 255) / 256), 256, 0, stream>>>(hbuf, ctxb, n);
    }

    // ---- decoder front ----
    embed_dec_kernel<<<(Mdec * DD + 255) / 256, 256, 0, stream>>>(sem_ids_fut, table, tte, bos, y0);
    rms(y0, norm_cw, dzn, Mdec, DD);
    gemm(dzn, DD, t_inc, nullptr, ybuf, nullptr, Mdec, AA, DD, 0);

    // ---- decoder layers ----
    for (int l = 0; l < LLAYERS; ++l) {
        const size_t wo = (size_t)l * SZ_AA;
        const size_t wf = (size_t)l * AA * DFF_C;
        rms(ybuf, g_ln1 + (size_t)l * AA, dzn, Mdec, AA);
        gemm(dzn, AA, tg_Wq + wo, nullptr, nullptr, dq, Mdec, AA, AA, 0);
        gemm(dzn, AA, tg_Wk + wo, nullptr, nullptr, dk, Mdec, AA, AA, 0);
        gemm(dzn, AA, tg_Wv + wo, nullptr, nullptr, dv, Mdec, AA, AA, 0);
        attn(dq, dk, dv, dob, nullptr, SID_C, SID_C, 1);
        gemm(dob, AA, tg_Wo + wo, ybuf, ybuf, nullptr, Mdec, AA, AA, 0);
        rms(ybuf, g_lnc + (size_t)l * AA, dzn, Mdec, AA);
        gemm(dzn, AA, tg_Cq + wo, nullptr, nullptr, dq, Mdec, AA, AA, 0);
        gemm(ctxb, AA, tg_Ck + wo, nullptr, nullptr, kb, Menc, AA, AA, 0);
        gemm(ctxb, AA, tg_Cv + wo, nullptr, nullptr, vb, Menc, AA, AA, 0);
        attn(dq, kb, vb, dob, maskw, SID_C, S_ENC_C, 0);
        gemm(dob, AA, tg_Co + wo, ybuf, ybuf, nullptr, Mdec, AA, AA, 0);
        rms(ybuf, g_ln2 + (size_t)l * AA, dzn, Mdec, AA);
        gemm(dzn, AA, tg_W1 + wf, nullptr, nullptr, dff, Mdec, DFF_C, AA, 1);
        gemm(dff, DFF_C, tg_W2 + wf, ybuf, ybuf, nullptr, Mdec, AA, DFF_C, 0);
    }

    // ---- logits ----
    {
        size_t n = (size_t)Mdec * AA;
        cvt_bf16_kernel<<<(unsigned)((n + 255) / 256), 256, 0, stream>>>(ybuf, ybf, n);
    }
    gemm(ybf, AA, t_out, nullptr, (float*)d_out, nullptr, Mdec, VV, AA, 0);
}
